// Encoder_69303592288576
// MI455X (gfx1250) — matrix operand order repaired and checked by reading, not compiled, not run on hardware
//
#include <hip/hip_runtime.h>
#include <hip/hip_bf16.h>

typedef __attribute__((ext_vector_type(16))) _Float16 v16h;
typedef __attribute__((ext_vector_type(8)))  _Float16 v8h;
typedef __attribute__((ext_vector_type(8)))  float    v8f;

#define WG 256

// ---- monotone float <-> uint encoding for atomic segment-max ---------------
__device__ __forceinline__ unsigned f32_to_ord(float f) {
  unsigned u = __float_as_uint(f);
  return (u & 0x80000000u) ? ~u : (u | 0x80000000u);
}
__device__ __forceinline__ float ord_to_f32(unsigned u) {
  return (u & 0x80000000u) ? __uint_as_float(u & 0x7fffffffu)
                           : __uint_as_float(~u);
}

// ---- f32 -> f16 staging copy ----------------------------------------------
__global__ void f32_to_f16_kernel(const float* __restrict__ in,
                                  _Float16* __restrict__ out, int n) {
  int i = blockIdx.x * WG + threadIdx.x;
  if (i < n) out[i] = (_Float16)in[i];
}

// ---- pack W[K,F] (f32 row-major) into CDNA5 WMMA B-fragment order ----------
// Layout: Wp[((kt*CT + ct)*32 + lane)*16 + i], where the fragment element i of
// lane holds W[kt*32 + (i/8)*16 + (lane/16)*8 + (i%8)][ct*16 + (lane%16)].
// Each lane's 16 halves become one contiguous 32-byte chunk.
__global__ void pack_w_kernel(const float* __restrict__ W,
                              _Float16* __restrict__ Wp, int K, int F) {
  int idx = blockIdx.x * WG + threadIdx.x;
  if (idx >= K * F) return;
  int i  = idx & 15;
  int l  = (idx >> 4) & 31;
  int t  = idx >> 9;            // (kt, ct) pair index; 512 elems per pair
  int CT = F >> 4;
  int ct = t % CT;
  int kt = t / CT;
  int krow = kt * 32 + (i >> 3) * 16 + (l >> 4) * 8 + (i & 7);
  int col  = ct * 16 + (l & 15);
  Wp[idx] = (_Float16)W[krow * F + col];
}

// ---- WMMA GEMM: H[N,F] = Xh[N,K] @ W, one wave per 16-row strip ------------
// A fragment loaded once per k-step, reused across all CT column tiles.
template <int CT>
__global__ __launch_bounds__(32) void gat_wmma_gemm(
    const _Float16* __restrict__ Xh,   // [N, K] row-major f16
    const _Float16* __restrict__ Wp,   // packed B fragments
    float* __restrict__ H,             // [N, CT*16] f32
    int K) {
  const int lane = threadIdx.x;        // 0..31, wave32 (EXEC all-ones)
  const int g    = lane >> 4;
  const int lr   = lane & 15;
  const int row0 = blockIdx.x * 16;
  const int F    = CT * 16;

  v8f acc[CT];
#pragma unroll
  for (int ct = 0; ct < CT; ++ct) acc[ct] = (v8f){};

  const int KT = K >> 5;
  for (int kt = 0; kt < KT; ++kt) {
    // A fragment (16x32 f16): lane lr holds row M=lr; per-lane data is two
    // contiguous 16-byte chunks -> two global_load_b128.
    const _Float16* arow = Xh + (size_t)(row0 + lr) * K + kt * 32 + g * 8;
    v8h alo = *(const v8h*)(arow);
    v8h ahi = *(const v8h*)(arow + 16);
    v16h a;
#pragma unroll
    for (int i = 0; i < 8; ++i) { a[i] = alo[i]; a[8 + i] = ahi[i]; }

    const _Float16* bbase = Wp + (size_t)kt * CT * 512 + (size_t)lane * 16;
#pragma unroll
    for (int ct = 0; ct < CT; ++ct) {
      v16h b = *(const v16h*)(bbase + (size_t)ct * 512);  // contiguous 32B
      acc[ct] = __builtin_amdgcn_wmma_f32_16x16x32_f16(
          false, a, false, b, (short)0, acc[ct], false, false);
    }
  }

  // D layout: VGPR r of lane -> (M = r + 8*g, N = lr)
#pragma unroll
  for (int ct = 0; ct < CT; ++ct) {
#pragma unroll
    for (int r = 0; r < 8; ++r) {
      H[(size_t)(row0 + r + 8 * g) * F + ct * 16 + lr] = acc[ct][r];
    }
  }
}

// ---- per-node: attention projections + init accumulators -------------------
__global__ void attn_init_kernel(const float* __restrict__ H,
                                 const float* __restrict__ a_src,
                                 const float* __restrict__ a_dst,
                                 int N, int F,
                                 float* __restrict__ als,
                                 float* __restrict__ ald,
                                 unsigned* __restrict__ nmax,
                                 float* __restrict__ nsum,
                                 float* __restrict__ acc) {
  int n = blockIdx.x * WG + threadIdx.x;
  if (n >= N) return;
  const float* h = H + (size_t)n * F;
  float s = 0.f, d = 0.f;
  for (int f = 0; f < F; ++f) {
    float v = h[f];
    s += v * a_src[f];
    d += v * a_dst[f];
  }
  als[n] = s;
  ald[n] = d;
  nmax[n] = 0u;          // identity for ordered-uint max over real floats
  nsum[n] = 0.f;
  float* ao = acc + (size_t)n * F;
  for (int f = 0; f < F; ++f) ao[f] = 0.f;
}

__device__ __forceinline__ void edge_sd(const long long* __restrict__ srcA,
                                        const long long* __restrict__ dstA,
                                        long long e, long long E0,
                                        int& s, int& d) {
  if (e < E0) {
    s = (int)srcA[e];
    d = (int)dstA[e];
  } else {  // appended self-loops
    s = d = (int)(e - E0);
  }
}

__device__ __forceinline__ float edge_logit(float as, float ad) {
  float z = as + ad;
  return z > 0.f ? z : 0.2f * z;  // leaky_relu, PyG default slope
}

// ---- pass 1: segment max over dst -----------------------------------------
__global__ void edge_max_kernel(const long long* __restrict__ srcA,
                                const long long* __restrict__ dstA,
                                long long E0, long long Etot,
                                const float* __restrict__ als,
                                const float* __restrict__ ald,
                                unsigned* __restrict__ nmax) {
  long long e = (long long)blockIdx.x * WG + threadIdx.x;
  if (e >= Etot) return;
  int s, d;
  edge_sd(srcA, dstA, e, E0, s, d);
  float z = edge_logit(als[s], ald[d]);
  atomicMax(&nmax[d], f32_to_ord(z));
}

// ---- pass 2: segment sum of exp(logit - max) ------------------------------
__global__ void edge_sum_kernel(const long long* __restrict__ srcA,
                                const long long* __restrict__ dstA,
                                long long E0, long long Etot,
                                const float* __restrict__ als,
                                const float* __restrict__ ald,
                                const unsigned* __restrict__ nmax,
                                float* __restrict__ nsum) {
  long long e = (long long)blockIdx.x * WG + threadIdx.x;
  if (e >= Etot) return;
  int s, d;
  edge_sd(srcA, dstA, e, E0, s, d);
  float z = edge_logit(als[s], ald[d]);
  float m = ord_to_f32(nmax[d]);
  atomicAdd(&nsum[d], __expf(z - m));
}

// ---- pass 3: weighted scatter-sum, one wave per edge ----------------------
__global__ void edge_aggr_kernel(const long long* __restrict__ srcA,
                                 const long long* __restrict__ dstA,
                                 long long E0, long long Etot,
                                 const float* __restrict__ als,
                                 const float* __restrict__ ald,
                                 const unsigned* __restrict__ nmax,
                                 const float* __restrict__ nsum,
                                 const float* __restrict__ hw,
                                 float* __restrict__ acc, int F) {
  long long tid = (long long)blockIdx.x * WG + threadIdx.x;
  long long e = tid >> 5;            // wave index == edge index
  int lane = threadIdx.x & 31;
  if (e >= Etot) return;
  int s, d;
  edge_sd(srcA, dstA, e, E0, s, d);
  float z = edge_logit(als[s], ald[d]);
  float m = ord_to_f32(nmax[d]);
  float coef = __expf(z - m) / (nsum[d] + 1e-16f);
  const float* hs = hw + (size_t)s * F;
  float* ao = acc + (size_t)d * F;
  for (int f = lane; f < F; f += 32) {
    atomicAdd(&ao[f], coef * hs[f]);   // native GLOBAL_ATOMIC_ADD_F32 in L2
  }
}

// ---- bias + optional ELU ---------------------------------------------------
__global__ void finalize_kernel(const float* __restrict__ acc,
                                const float* __restrict__ bias,
                                float* __restrict__ dst,
                                long long total, int F, int apply_elu) {
  long long i = (long long)blockIdx.x * WG + threadIdx.x;
  if (i >= total) return;
  int f = (int)(i % F);
  float v = acc[i] + bias[f];
  if (apply_elu) v = v > 0.f ? v : (__expf(v) - 1.f);
  dst[i] = v;
}

extern "C" void kernel_launch(void* const* d_in, const int* in_sizes, int n_in,
                              void* d_out, int out_size, void* d_ws,
                              size_t ws_size, hipStream_t stream) {
  (void)n_in; (void)out_size; (void)ws_size;
  const float* x = (const float*)d_in[0];
  const long long* ei = (const long long*)d_in[1];  // int64 edge_index [2,E0]
  // d_in[2] = edge_attr: ignored by GATConv with edge_dim=None.

  const int N = in_sizes[0] / 64;          // 50000 (= 3125 * 16 exactly)
  const long long E0 = in_sizes[1] / 2;    // 800000
  const long long Etot = E0 + N;           // + self loops
  const long long* srcA = ei;
  const long long* dstA = ei + E0;

  static const int FIN[6]  = {64, 32, 64, 96, 128, 128};
  static const int FOUT[6] = {32, 64, 96, 128, 32, 32};

  // ---- workspace carve-up (256B aligned) ----
  char* ws = (char*)d_ws;
  auto carve = [&](size_t bytes) -> char* {
    char* p = ws;
    ws += (bytes + 255) & ~(size_t)255;
    return p;
  };
  float*    bufA = (float*)carve((size_t)N * 128 * sizeof(float));
  float*    bufB = (float*)carve((size_t)N * 128 * sizeof(float));
  float*    hw   = (float*)carve((size_t)N * 128 * sizeof(float));
  _Float16* xh   = (_Float16*)carve((size_t)N * 128 * sizeof(_Float16));
  _Float16* wp   = (_Float16*)carve((size_t)128 * 128 * sizeof(_Float16));
  float*    als  = (float*)carve((size_t)N * sizeof(float));
  float*    ald  = (float*)carve((size_t)N * sizeof(float));
  unsigned* nmax = (unsigned*)carve((size_t)N * sizeof(unsigned));
  float*    nsum = (float*)carve((size_t)N * sizeof(float));

  // layer-0 input = x
  hipMemcpyAsync(bufA, x, (size_t)N * 64 * sizeof(float),
                 hipMemcpyDeviceToDevice, stream);

  float* cur = bufA;
  float* nxt = bufB;
  for (int L = 0; L < 6; ++L) {
    const int Fi = FIN[L], Fo = FOUT[L];
    const float* W  = (const float*)d_in[3 + 4 * L + 0];
    const float* as = (const float*)d_in[3 + 4 * L + 1];
    const float* ad = (const float*)d_in[3 + 4 * L + 2];
    const float* bb = (const float*)d_in[3 + 4 * L + 3];

    // stage f16 activations + packed f16 B-fragments
    int nx = N * Fi;
    f32_to_f16_kernel<<<(nx + WG - 1) / WG, WG, 0, stream>>>(cur, xh, nx);
    int nw = Fi * Fo;
    pack_w_kernel<<<(nw + WG - 1) / WG, WG, 0, stream>>>(W, wp, Fi, Fo);

    // dense transform: h = cur @ W  (one wave per 16-row strip, CT col tiles)
    unsigned gR = (unsigned)(N / 16);
    switch (Fo >> 4) {
      case 2: gat_wmma_gemm<2><<<gR, 32, 0, stream>>>(xh, wp, hw, Fi); break;
      case 4: gat_wmma_gemm<4><<<gR, 32, 0, stream>>>(xh, wp, hw, Fi); break;
      case 6: gat_wmma_gemm<6><<<gR, 32, 0, stream>>>(xh, wp, hw, Fi); break;
      default: gat_wmma_gemm<8><<<gR, 32, 0, stream>>>(xh, wp, hw, Fi); break;
    }

    // layers 0-3 accumulate into the ping-pong target; layers 4/5 reuse the
    // buffer that is NOT the (preserved) layer-3 output.
    float* acc = (L < 4) ? nxt : ((cur == bufA) ? bufB : bufA);

    attn_init_kernel<<<(N + WG - 1) / WG, WG, 0, stream>>>(
        hw, as, ad, N, Fo, als, ald, nmax, nsum, acc);

    unsigned gE = (unsigned)((Etot + WG - 1) / WG);
    edge_max_kernel<<<gE, WG, 0, stream>>>(srcA, dstA, E0, Etot, als, ald, nmax);
    edge_sum_kernel<<<gE, WG, 0, stream>>>(srcA, dstA, E0, Etot, als, ald,
                                           nmax, nsum);
    unsigned gA = (unsigned)((Etot * 32 + WG - 1) / WG);
    edge_aggr_kernel<<<gA, WG, 0, stream>>>(srcA, dstA, E0, Etot, als, ald,
                                            nmax, nsum, hw, acc, Fo);

    float* dst = (L < 4) ? nxt
               : (L == 4) ? (float*)d_out
                          : (float*)d_out + (size_t)N * 32;
    long long total = (long long)N * Fo;
    finalize_kernel<<<(unsigned)((total + WG - 1) / WG), WG, 0, stream>>>(
        acc, bb, dst, total, Fo, (L < 4) ? 1 : 0);

    if (L < 4) { float* t = cur; cur = nxt; nxt = t; }
  }
}
